// Cross_attention_89730456748795
// MI455X (gfx1250) — matrix operand order repaired and checked by reading, not compiled, not run on hardware
//
#include <hip/hip_runtime.h>
#include <hip/hip_bf16.h>
#include <math.h>

// ---------------------------------------------------------------------------
// Cross-attention block for MI455X (gfx1250, wave32, WMMA).
// All matmuls: v_wmma_f32_16x16x32_bf16 (bf16 in, f32 accumulate).
// Flash attention: 16-query x 32-key tiles, online softmax; double-buffered
// K/V tiles staged global->LDS with GLOBAL_LOAD_ASYNC_TO_LDS_B128 (ASYNCcnt),
// overlapping DMA of chunk i+1 with the WMMAs of chunk i. All fragments are
// contiguous-per-lane (b128 / ds_load_b128).
// ---------------------------------------------------------------------------

typedef __attribute__((ext_vector_type(16))) __bf16 v16bf;
typedef __attribute__((ext_vector_type(8)))  __bf16 v8bf;
typedef __attribute__((ext_vector_type(8)))  float  v8f;
typedef __attribute__((ext_vector_type(4)))  int    v4i;

#if defined(__gfx1250__) && \
    __has_builtin(__builtin_amdgcn_global_load_async_to_lds_b128) && \
    __has_builtin(__builtin_amdgcn_s_wait_asynccnt)
#define ASYNC_LDS 1
#else
#define ASYNC_LDS 0
#endif

// 16-byte global -> LDS copy: async (no VGPR round trip, ASYNCcnt) if available
__device__ __forceinline__ void cp16_lds(const __bf16* g, __bf16* l) {
#if ASYNC_LDS
  __builtin_amdgcn_global_load_async_to_lds_b128(
      (__attribute__((address_space(1))) v4i*)(uintptr_t)g,
      (__attribute__((address_space(3))) v4i*)(uintptr_t)l, 0, 0);
#else
  *(uint4*)l = *(const uint4*)g;
#endif
}
__device__ __forceinline__ void cp_wait_all() {
#if ASYNC_LDS
  __builtin_amdgcn_s_wait_asynccnt(0);
#endif
}

// ---- WMMA fragment builders (wave32; this code's contiguous k order, same for A and B) ----
// A 16x32 (MxK bf16): lanes 0-15: M=lane, K=elem
//                     lanes 16-31: M=lane-16, K=16+elem
template <typename F>
__device__ __forceinline__ v16bf mk_afrag(int lane, F f) {
  v16bf a;
  int m  = lane & 15;
  int kb = (lane & 16) ? 16 : 0;
#pragma unroll
  for (int e = 0; e < 16; ++e) {
    int k = kb + e;
    a[e] = f(m, k);
  }
  return a;
}
// B 32x16 (KxN bf16): lanes 0-15: N=lane, K=elem; lanes 16-31: K=16+elem
template <typename F>
__device__ __forceinline__ v16bf mk_bfrag(int lane, F f) {
  v16bf b;
  int n  = lane & 15;
  int kb = (lane & 16) ? 16 : 0;
#pragma unroll
  for (int e = 0; e < 16; ++e) b[e] = f(kb + e, n);
  return b;
}
// C/D 16x16 f32: VGPR r, lanes 0-15 -> (M=r, N=lane); lanes 16-31 -> (M=8+r,N=lane-16)

__device__ __forceinline__ v8f wmma_bf16(v16bf a, v16bf b, v8f c) {
  return __builtin_amdgcn_wmma_f32_16x16x32_bf16(false, a, false, b, (short)0, c,
                                                 false, false);
}

__device__ __forceinline__ int reflect64(int i) {
  return i < 0 ? -i : (i > 63 ? 126 - i : i);
}

// ---------------------------------------------------------------------------
// Kernel 1: both depthwise convs (3x3 pad1, 5x5 pad2), reflect pad, LeakyReLU
// ---------------------------------------------------------------------------
__global__ __launch_bounds__(256) void dwconv_kernel(
    const float* __restrict__ in, const float* __restrict__ w1, const float* __restrict__ b1,
    const float* __restrict__ w2, const float* __restrict__ b2,
    float* __restrict__ xo, float* __restrict__ yo)
{
  int idx = blockIdx.x * 256 + threadIdx.x;          // B*C*H*W = 1<<20
  int wp = idx & 63, hp = (idx >> 6) & 63, c = (idx >> 12) & 63;
  const float* ip = in + ((idx >> 12) << 12);
  float s1 = 0.f;
#pragma unroll
  for (int dy = -1; dy <= 1; ++dy)
#pragma unroll
    for (int dx = -1; dx <= 1; ++dx)
      s1 += ip[reflect64(hp + dy) * 64 + reflect64(wp + dx)] *
            w1[c * 9 + (dy + 1) * 3 + (dx + 1)];
  s1 += b1[c];
  xo[idx] = s1 > 0.f ? s1 : 0.01f * s1;

  float s2 = 0.f;
#pragma unroll
  for (int dy = -2; dy <= 2; ++dy)
#pragma unroll
    for (int dx = -2; dx <= 2; ++dx)
      s2 += ip[reflect64(hp + dy) * 64 + reflect64(wp + dx)] *
            w2[c * 25 + (dy + 2) * 5 + (dx + 2)];
  s2 += b2[c];
  yo[idx] = s2 > 0.f ? s2 : 0.01f * s2;
}

// ---------------------------------------------------------------------------
// Kernel 2: GroupNorm statistics. blk = branch*64 + b*16 + g   (128 blocks)
// ---------------------------------------------------------------------------
__global__ __launch_bounds__(256) void gn_stats_kernel(
    const float* __restrict__ x, const float* __restrict__ y, float* __restrict__ stats)
{
  int blk = blockIdx.x;
  const float* src = (blk & 64) ? y : x;
  int b = (blk >> 4) & 3, g = blk & 15;
  const float* p = src + ((b * 64 + g * 4) << 12);
  float s = 0.f, s2 = 0.f;
  for (int i = threadIdx.x; i < 16384; i += 256) { float v = p[i]; s += v; s2 += v * v; }
  __shared__ float shs[256], shq[256];
  shs[threadIdx.x] = s; shq[threadIdx.x] = s2;
  __syncthreads();
  for (int off = 128; off > 0; off >>= 1) {
    if (threadIdx.x < off) {
      shs[threadIdx.x] += shs[threadIdx.x + off];
      shq[threadIdx.x] += shq[threadIdx.x + off];
    }
    __syncthreads();
  }
  if (threadIdx.x == 0) {
    float mu  = shs[0] * (1.f / 16384.f);
    float var = shq[0] * (1.f / 16384.f) - mu * mu;
    stats[blk * 2]     = mu;
    stats[blk * 2 + 1] = rsqrtf(var + 1e-5f);
  }
}

// ---------------------------------------------------------------------------
// Kernel 3: apply GroupNorm -> f32 copy (residual, [c][hw]) and bf16 copy
// TRANSPOSED ([hw][64]) so the qkv GEMM B-fragment is contiguous per lane.
// ---------------------------------------------------------------------------
__global__ __launch_bounds__(256) void gn_apply_kernel(
    const float* __restrict__ x, const float* __restrict__ y, const float* __restrict__ stats,
    const float* __restrict__ gAw, const float* __restrict__ gAb,
    const float* __restrict__ gBw, const float* __restrict__ gBb,
    float* __restrict__ xAn, float* __restrict__ xBn,
    __bf16* __restrict__ xAt, __bf16* __restrict__ xBt)
{
  int idx = blockIdx.x * 256 + threadIdx.x;          // 2 * (1<<20)
  int branch = idx >> 20;
  int rem = idx & ((1 << 20) - 1);
  int b = rem >> 18, c = (rem >> 12) & 63, hw = rem & 4095;
  int blk = branch * 64 + b * 16 + (c >> 2);
  float mu = stats[blk * 2], rstd = stats[blk * 2 + 1];
  float v  = (branch ? y : x)[rem];
  float wv = (branch ? gBw : gAw)[c];
  float bv = (branch ? gBb : gAb)[c];
  float r  = (v - mu) * rstd * wv + bv;
  int tidx = b * 262144 + hw * 64 + c;               // [b][hw][c]
  if (branch) { xBn[rem] = r; xBt[tidx] = (__bf16)r; }
  else        { xAn[rem] = r; xAt[tidx] = (__bf16)r; }
}

// ---------------------------------------------------------------------------
// Kernel 4: qkv GEMM [192x64]@[64x4096] per (batch,branch).
// Outputs per (b,branch) section of 3*262144 bf16:
//   Qt [pixel][64]  (pre-scaled by 1/8),  Kt [pixel][64],  V [d][pixel]
// Qt/Kt written as packed 16B v8bf stores (8 consecutive channels per lane).
// ---------------------------------------------------------------------------
__global__ __launch_bounds__(256) void qkv_gemm_kernel(
    const __bf16* __restrict__ xAt, const __bf16* __restrict__ xBt,
    const float* __restrict__ wA, const float* __restrict__ wB,
    __bf16* __restrict__ qkvA, __bf16* __restrict__ qkvB)
{
  int wid  = blockIdx.x * 8 + (threadIdx.x >> 5);
  int lane = threadIdx.x & 31;
  int b = wid / 6144, rem = wid % 6144;
  int br = rem / 3072, t = rem % 3072;
  int m0 = (t >> 8) * 16, n0 = (t & 255) * 16;
  const __bf16* X = (br ? xBt : xAt) + b * 262144;   // [hw][64]
  const float*  W = br ? wB : wA;
  __bf16* sec = (br ? qkvB : qkvA) + b * 786432;
  v8f acc = {};
#pragma unroll
  for (int dc = 0; dc < 2; ++dc) {
    v16bf a  = mk_afrag(lane, [&](int m, int k) -> __bf16 {
      return (__bf16)W[(m0 + m) * 64 + dc * 32 + k]; });
    v16bf bb = mk_bfrag(lane, [&](int k, int n) -> __bf16 {
      return X[(n0 + n) * 64 + dc * 32 + k]; });     // contiguous per lane
    acc = wmma_bf16(a, bb, acc);
  }
  int half8 = (lane & 16) ? 8 : 0;
  int col = n0 + (lane & 15);
  if (m0 < 128) {                                    // q or k -> transposed layout
    float scale = (m0 < 64) ? 0.125f : 1.f;
    __bf16* dst = sec + (m0 < 64 ? 0 : 262144) + col * 64 + (m0 & 63) + half8;
    v8bf pk;
#pragma unroll
    for (int r = 0; r < 8; ++r) pk[r] = (__bf16)(acc[r] * scale);
    *(v8bf*)dst = pk;                                // 16B aligned packed store
  } else {                                           // v -> [d][pixel]
    __bf16* dst = sec + 524288;
#pragma unroll
    for (int r = 0; r < 8; ++r)
      dst[((m0 - 128 + half8 + r) << 12) + col] = (__bf16)acc[r];
  }
}

// ---------------------------------------------------------------------------
// Kernel 5: flash attention. One block = 8 waves = 8 query tiles of the SAME
// (batch, branch). K/V 32-key tiles are double-buffered in LDS: chunk i+1 is
// issued with async global->LDS copies (ASYNCcnt) before computing chunk i,
// so the DMA overlaps 8 WMMAs + softmax; one s_wait_asynccnt + one barrier
// per iteration. Prefetch warms L2 two chunks ahead (global_prefetch_b8).
// Output: At [query][64] (transposed) for the out-proj B-fragment.
// ---------------------------------------------------------------------------
__global__ __launch_bounds__(256) void attn_kernel(
    const __bf16* __restrict__ qkvA, const __bf16* __restrict__ qkvB,
    __bf16* __restrict__ AtA, __bf16* __restrict__ AtB)
{
  __shared__ __align__(16) __bf16 kt_lds[2][32 * 64];  // [buf][key rel][d]
  __shared__ __align__(16) __bf16 vt_lds[2][64 * 32];  // [buf][d][key rel]
  __shared__ __align__(16) __bf16 p_lds[8][512];       // 16x32 P per wave
  int tid = threadIdx.x;
  int w8 = tid >> 5, lane = tid & 31;
  int blk = blockIdx.x;                                // 256 blocks
  int b = blk >> 6, r6 = blk & 63;
  int br = r6 >> 5, qg = r6 & 31;
  const __bf16* baseq  = (br ? qkvA : qkvB) + b * 786432;   // queries
  const __bf16* basekv = (br ? qkvB : qkvA) + b * 786432;   // keys/values
  const __bf16* Qt = baseq;                            // [pixel][64], pre-scaled
  const __bf16* Kt = basekv + 262144;                  // [pixel][64]
  const __bf16* V  = basekv + 524288;                  // [d][pixel]
  __bf16* At = (br ? AtB : AtA) + b * 262144;          // [query][64]
  int q0 = (qg * 8 + w8) << 4;

  v16bf qa[2];
#pragma unroll
  for (int dc = 0; dc < 2; ++dc)
    qa[dc] = mk_afrag(lane, [&](int m, int k) -> __bf16 {
      return Qt[(q0 + m) * 64 + dc * 32 + k]; });      // contiguous per lane

  float mrow[8], lrow[8];
  v8f o[4] = {};
#pragma unroll
  for (int r = 0; r < 8; ++r) { mrow[r] = -1e30f; lrow[r] = 0.f; }

  // this thread's 16B staging slice
  int krow = tid >> 3, koff = (tid & 7) * 8;           // 32 rows x 128B
  int vrow = tid >> 2, voff = (tid & 3) * 8;           // 64 rows x 64B

  auto stage = [&](int buf, int kc) {
    int k0 = kc << 5;
    cp16_lds(&Kt[(k0 + krow) * 64 + koff], &kt_lds[buf][krow * 64 + koff]);
    cp16_lds(&V[(vrow << 12) + k0 + voff], &vt_lds[buf][vrow * 32 + voff]);
  };

  stage(0, 0);                                         // prologue fill
  for (int kc = 0; kc < 128; ++kc) {
    int cur = kc & 1;
    cp_wait_all();                                     // buf `cur` resident
    __syncthreads();                                   // all waves done w/ 1-cur
    if (kc + 1 < 128) stage(cur ^ 1, kc + 1);          // overlap next-chunk DMA
    if (kc + 2 < 128) {                                // L2 warm 2 chunks ahead
      int kp = (kc + 2) << 5;
      __builtin_prefetch(&Kt[(kp + krow) * 64 + koff], 0, 1);
      __builtin_prefetch(&V[(vrow << 12) + kp + voff], 0, 1);
    }
    const __bf16* kt = kt_lds[cur];
    const __bf16* vt = vt_lds[cur];

    v8f s0 = {}, s1 = {};
#pragma unroll
    for (int dc = 0; dc < 2; ++dc) {
      v16bf kb0 = mk_bfrag(lane, [&](int k, int n) -> __bf16 {
        return kt[n * 64 + dc * 32 + k]; });
      v16bf kb1 = mk_bfrag(lane, [&](int k, int n) -> __bf16 {
        return kt[(16 + n) * 64 + dc * 32 + k]; });
      s0 = wmma_bf16(qa[dc], kb0, s0);
      s1 = wmma_bf16(qa[dc], kb1, s1);
    }
    float corr[8];
#pragma unroll
    for (int r = 0; r < 8; ++r) {                      // online softmax per row
      float mx = fmaxf(s0[r], s1[r]);
      mx = fmaxf(mx, __shfl_xor(mx, 1, 32));
      mx = fmaxf(mx, __shfl_xor(mx, 2, 32));
      mx = fmaxf(mx, __shfl_xor(mx, 4, 32));
      mx = fmaxf(mx, __shfl_xor(mx, 8, 32));           // stays in 16-lane half
      float nm = fmaxf(mrow[r], mx);
      float c  = __expf(mrow[r] - nm);
      float p0 = __expf(s0[r] - nm);
      float p1 = __expf(s1[r] - nm);
      float rs = p0 + p1;
      rs += __shfl_xor(rs, 1, 32);
      rs += __shfl_xor(rs, 2, 32);
      rs += __shfl_xor(rs, 4, 32);
      rs += __shfl_xor(rs, 8, 32);
      lrow[r] = lrow[r] * c + rs;
      mrow[r] = nm;
      corr[r] = c;
      s0[r] = p0;
      s1[r] = p1;
    }
#pragma unroll
    for (int dt = 0; dt < 4; ++dt)
#pragma unroll
      for (int r = 0; r < 8; ++r) o[dt][r] *= corr[r];

    int rowb = (lane & 16) ? 8 : 0;                    // P: C-layout -> LDS
#pragma unroll
    for (int r = 0; r < 8; ++r) {
      p_lds[w8][(rowb + r) * 32 + (lane & 15)]      = (__bf16)s0[r];
      p_lds[w8][(rowb + r) * 32 + (lane & 15) + 16] = (__bf16)s1[r];
    }
    v16bf pa = mk_afrag(lane, [&](int m, int k) -> __bf16 {
      return p_lds[w8][m * 32 + k]; });
#pragma unroll
    for (int dt = 0; dt < 4; ++dt) {
      v16bf vb = mk_bfrag(lane, [&](int k, int n) -> __bf16 {
        return vt[(dt * 16 + n) * 32 + k]; });         // contiguous per lane
      o[dt] = wmma_bf16(pa, vb, o[dt]);
    }
  }
  int rowb = (lane & 16) ? 8 : 0;
#pragma unroll
  for (int r = 0; r < 8; ++r) {
    float inv = 1.f / lrow[r];
#pragma unroll
    for (int dt = 0; dt < 4; ++dt)                     // At[query][d]
      At[(q0 + rowb + r) * 64 + dt * 16 + (lane & 15)] = (__bf16)(o[dt][r] * inv);
  }
}

// ---------------------------------------------------------------------------
// Kernel 6: out projection [64x64]@[64x4096] + bias + residual, concat
// ---------------------------------------------------------------------------
__global__ __launch_bounds__(256) void out_gemm_kernel(
    const __bf16* __restrict__ AtA, const __bf16* __restrict__ AtB,
    const float* __restrict__ wA, const float* __restrict__ bA,
    const float* __restrict__ wB, const float* __restrict__ bB,
    const float* __restrict__ xAn, const float* __restrict__ xBn,
    float* __restrict__ out)
{
  int wid  = blockIdx.x * 8 + (threadIdx.x >> 5);
  int lane = threadIdx.x & 31;
  int b = wid / 2048, rem = wid % 2048;
  int br = rem / 1024, t = rem % 1024;
  int m0 = (t >> 8) * 16, n0 = (t & 255) * 16;
  const __bf16* At = (br ? AtB : AtA) + b * 262144;    // [query][64]
  const float* Wt   = br ? wB : wA;
  const float* bias = br ? bB : bA;
  const float* res  = (br ? xBn : xAn) + (b << 18);
  v8f acc = {};
#pragma unroll
  for (int dc = 0; dc < 2; ++dc) {
    v16bf a  = mk_afrag(lane, [&](int m, int k) -> __bf16 {
      return (__bf16)Wt[(m0 + m) * 64 + dc * 32 + k]; });
    v16bf bb = mk_bfrag(lane, [&](int k, int n) -> __bf16 {
      return At[(n0 + n) * 64 + dc * 32 + k]; });      // contiguous per lane
    acc = wmma_bf16(a, bb, acc);
  }
  int half8 = (lane & 16) ? 8 : 0;
#pragma unroll
  for (int r = 0; r < 8; ++r) {
    int row = m0 + half8 + r;
    int col = n0 + (lane & 15);
    out[((b * 128 + br * 64 + row) << 12) + col] =
        acc[r] + bias[row] + res[(row << 12) + col];
  }
}

// ---------------------------------------------------------------------------
extern "C" void kernel_launch(void* const* d_in, const int* in_sizes, int n_in,
                              void* d_out, int out_size, void* d_ws, size_t ws_size,
                              hipStream_t stream)
{
  const float* inp   = (const float*)d_in[0];
  const float* dw1w  = (const float*)d_in[1];
  const float* dw1b  = (const float*)d_in[2];
  const float* dw2w  = (const float*)d_in[3];
  const float* dw2b  = (const float*)d_in[4];
  const float* gAw   = (const float*)d_in[5];
  const float* gAb   = (const float*)d_in[6];
  const float* gBw   = (const float*)d_in[7];
  const float* gBb   = (const float*)d_in[8];
  const float* qkvAw = (const float*)d_in[9];
  const float* qkvBw = (const float*)d_in[10];
  const float* outAw = (const float*)d_in[11];
  const float* outAb = (const float*)d_in[12];
  const float* outBw = (const float*)d_in[13];
  const float* outBb = (const float*)d_in[14];

  // workspace layout (~36 MB)
  float* x     = (float*)d_ws;              // [4,64,64,64]
  float* y     = x   + (1 << 20);
  float* xAn   = y   + (1 << 20);           // normalized residual (f32, [c][hw])
  float* xBn   = xAn + (1 << 20);
  float* stats = xBn + (1 << 20);           // 256 used, 1024 reserved
  __bf16* xAt  = (__bf16*)(stats + 1024);   // normalized bf16, [hw][64]
  __bf16* xBt  = xAt + (1 << 20);
  __bf16* qkvA = xBt + (1 << 20);           // per b: Qt | Kt | V   (3*262144)
  __bf16* qkvB = qkvA + 786432 * 4;
  __bf16* AtA  = qkvB + 786432 * 4;         // attention out, [query][64]
  __bf16* AtB  = AtA + 262144 * 4;

  dwconv_kernel  <<<4096, 256, 0, stream>>>(inp, dw1w, dw1b, dw2w, dw2b, x, y);
  gn_stats_kernel<<< 128, 256, 0, stream>>>(x, y, stats);
  gn_apply_kernel<<<8192, 256, 0, stream>>>(x, y, stats, gAw, gAb, gBw, gBb,
                                            xAn, xBn, xAt, xBt);
  qkv_gemm_kernel<<<3072, 256, 0, stream>>>(xAt, xBt, qkvAw, qkvBw, qkvA, qkvB);
  attn_kernel    <<< 256, 256, 0, stream>>>(qkvA, qkvB, AtA, AtB);
  out_gemm_kernel<<<1024, 256, 0, stream>>>(AtA, AtB, outAw, outAb, outBw, outBb,
                                            xAn, xBn, (float*)d_out);
}
